// ReservoirLayer_77446850282066
// MI455X (gfx1250) — compile-verified, matrix operand order checked by reading, not run on hardware
//
#include <hip/hip_runtime.h>
#include <hip/hip_bf16.h>

// Problem constants (from reference)
#define BATCH 32
#define SEQ   2048
#define IDIM  64
#define RES   1024
#define LEAK  0.3f

// Scan config: 32 persistent WGs x 128 threads (4 waves).
// wave = {Mhalf: batch 0-15 / 16-31} x {Ntile: which 16-column tile}
#define NWG    32
#define RTILE  32
#define BLOCK  128

typedef __attribute__((ext_vector_type(16))) __bf16 v16bf;
typedef __attribute__((ext_vector_type(8)))  float  v8f;

union Frag { v16bf v; uint4 q[2]; };

// round-to-nearest-even f32 -> bf16 bits
__device__ __forceinline__ unsigned short f2bf(float f) {
  unsigned u = __float_as_uint(f);
  unsigned r = u + 0x7FFFu + ((u >> 16) & 1u);
  return (unsigned short)(r >> 16);
}
__device__ __forceinline__ unsigned packbf(float a, float b) {
  return (unsigned)f2bf(a) | ((unsigned)f2bf(b) << 16);
}
// branch-free tanh: 1 - 2/(exp(2x)+1), via v_exp_f32 + v_rcp_f32
__device__ __forceinline__ float fast_tanh(float x) {
  const float xc = fminf(fmaxf(x, -10.0f), 10.0f);
  const float p  = __builtin_amdgcn_exp2f(xc * 2.8853900817779268f); // 2*log2(e)
  return 1.0f - 2.0f * __builtin_amdgcn_rcpf(p + 1.0f);
}

// ---------------- Phase 1: u = x @ W_in.T, in-place into d_out ----------------
__global__ void __launch_bounds__(256)
esn_input_gemm(const float* __restrict__ x,
               const float* __restrict__ Win,
               float* __restrict__ out) {
  __shared__ float xs[IDIM];
  const int bt  = blockIdx.x;                       // b*T + t
  const int col = blockIdx.y * 256 + threadIdx.x;   // reservoir column
  if (threadIdx.x < IDIM) xs[threadIdx.x] = x[(size_t)bt * IDIM + threadIdx.x];
  __syncthreads();
  const float* w = Win + (size_t)col * IDIM;
  float acc = 0.f;
#pragma unroll
  for (int i = 0; i < IDIM; ++i) acc = fmaf(xs[i], w[i], acc);
  out[(size_t)bt * RES + col] = acc;
}

// ---------------- Phase 2: persistent recurrent scan ----------------
// C[b,r] = u[b,t,r] + sum_k Sbf[b,k] * W_res[r,k]  (bf16 WMMA, f32 accum)
__global__ void __launch_bounds__(BLOCK)
esn_scan(const float* __restrict__ Wres,
         float* __restrict__ out,
         unsigned short* __restrict__ Sb0,   // bf16 state ping
         unsigned short* __restrict__ Sb1,   // bf16 state pong
         unsigned* __restrict__ barrier_ctr) {
  // W_res slice, bf16, pre-swizzled into per-lane WMMA B fragments:
  // [kk:32][ntile:2][lane:32][2 x uint4] = 64 KB
  __shared__ uint4 Blds[32 * 2 * 32 * 2];

  const int tid  = threadIdx.x;
  const int lane = tid & 31;
  const int wave = tid >> 5;
  const int c0   = blockIdx.x * RTILE;

  // ---- one-time: convert + swizzle W_res slice (contiguous 8-float reads) ----
  for (int j = tid; j < 32 * 2 * 32 * 2; j += BLOCK) {
    const int qi   = j & 1;           // low/high half of fragment (v0-3 / v4-7)
    const int l    = (j >> 1) & 31;
    const int nt_  = (j >> 6) & 1;
    const int kk   = j >> 7;
    const int colg = c0 + nt_ * 16 + (l & 15);
    const int kbase = kk * 32 + qi * 16 + (l >> 4) * 8;
    const float* wr = Wres + (size_t)colg * RES + kbase;
    uint4 q;
    q.x = packbf(wr[0], wr[1]);
    q.y = packbf(wr[2], wr[3]);
    q.z = packbf(wr[4], wr[5]);
    q.w = packbf(wr[6], wr[7]);
    Blds[j] = q;
  }
  __syncthreads();

  const int Moff = (wave & 1) * 16;
  const int nt   = wave >> 1;
  const int n_g  = c0 + nt * 16 + (lane & 15);
  const int half = lane >> 4;
  const int m    = Moff + (lane & 15);   // this lane's A-fragment batch row

  // per-lane epilogue coordinates are time-invariant -> keep prev state in regs
  float  prev[8];
  size_t uoff[8];   // offset of (b_g, t=0, n_g) in out, advance by t*RES
  int    soff[8];   // offset of (b_g, n_g) in bf16 state matrix
#pragma unroll
  for (int v = 0; v < 8; ++v) {
    const int b_g = Moff + v + 8 * half;
    prev[v] = 0.f;
    uoff[v] = ((size_t)b_g * SEQ) * RES + n_g;
    soff[v] = b_g * RES + n_g;
  }

  const uint4* bfragp = &Blds[(nt * 32 + lane) * 2];  // + kk*128 per k-step

  for (int t = 0; t < SEQ; ++t) {
    // state(t) lives in buf[t&1]; read state(t-1), write state(t)
    const unsigned short* SbR = (t & 1) ? Sb0 : Sb1;
    unsigned short*       SbW = (t & 1) ? Sb1 : Sb0;

    // C init = u tile (C/D layout); second accumulator breaks the D->C chain
    v8f c0v;
#pragma unroll
    for (int v = 0; v < 8; ++v) c0v[v] = out[uoff[v] + (size_t)t * RES];
    v8f c1v = {};

    if (t > 0) {
      const unsigned short* srow = SbR + m * RES;   // row-major bf16 state row
#pragma unroll 4
      for (int kk = 0; kk < 32; kk += 2) {
        Frag a0, b0, a1, b1;
        const unsigned short* s0 = srow + kk * 32 + half * 8;
        a0.q[0] = *(const uint4*)(s0);        // k: kk*32+half*8 .. +7
        a0.q[1] = *(const uint4*)(s0 + 16);   // k: kk*32+16+half*8 .. +7
        const uint4* bp0 = bfragp + (size_t)kk * 128;
        b0.q[0] = bp0[0];
        b0.q[1] = bp0[1];
        const unsigned short* s1 = s0 + 32;
        a1.q[0] = *(const uint4*)(s1);
        a1.q[1] = *(const uint4*)(s1 + 16);
        const uint4* bp1 = bp0 + 128;
        b1.q[0] = bp1[0];
        b1.q[1] = bp1[1];
        c0v = __builtin_amdgcn_wmma_f32_16x16x32_bf16(
                  false, a0.v, false, b0.v, (short)0, c0v, false, false);
        c1v = __builtin_amdgcn_wmma_f32_16x16x32_bf16(
                  false, a1.v, false, b1.v, (short)0, c1v, false, false);
      }
    }

    // epilogue: tanh + leaky update; write fp32 output + bf16 shadow state
#pragma unroll
    for (int v = 0; v < 8; ++v) {
      const float nv  = fast_tanh(c0v[v] + c1v[v]);
      const float res = (1.0f - LEAK) * prev[v] + LEAK * nv;
      prev[v] = res;
      const size_t o = uoff[v] + (size_t)t * RES;
      out[o] = res;
      SbW[soff[v]] = f2bf(res);
      if (t + 1 < SEQ) __builtin_prefetch(&out[o + RES], 0, 1); // next u slice
    }

    // ---- device-wide barrier (monotonic counter; release/acquire) ----
    __threadfence();
    __syncthreads();
    if (tid == 0) {
      atomicAdd(barrier_ctr, 1u);
      const unsigned target = (unsigned)(NWG * (t + 1));
      while (__hip_atomic_load(barrier_ctr, __ATOMIC_ACQUIRE,
                               __HIP_MEMORY_SCOPE_AGENT) < target) {
        __builtin_amdgcn_s_sleep(1);
      }
    }
    __syncthreads();
    __threadfence();
  }
}

extern "C" void kernel_launch(void* const* d_in, const int* in_sizes, int n_in,
                              void* d_out, int out_size, void* d_ws, size_t ws_size,
                              hipStream_t stream) {
  const float* x    = (const float*)d_in[0];   // [32, 2048, 64]
  const float* Win  = (const float*)d_in[1];   // [1024, 64]
  const float* Wres = (const float*)d_in[2];   // [1024, 1024]
  float*       out  = (float*)d_out;           // [32, 2048, 1024]

  unsigned*       ctr = (unsigned*)d_ws;
  unsigned short* Sb0 = (unsigned short*)((char*)d_ws + 4096);
  unsigned short* Sb1 = (unsigned short*)((char*)d_ws + 4096 + BATCH * RES * 2);

  // zero the global barrier counter every launch (capturable memset node)
  hipMemsetAsync(d_ws, 0, 256, stream);

  // Phase 1: u into d_out
  dim3 g1(BATCH * SEQ, RES / 256);
  esn_input_gemm<<<g1, 256, 0, stream>>>(x, Win, out);

  // Phase 2: persistent cooperative scan (32 co-resident WGs, 64 KB LDS each)
  esn_scan<<<NWG, BLOCK, 0, stream>>>(Wres, out, Sb0, Sb1, ctr);
}
